// CrossTransformerBlock3D_46454366274130
// MI455X (gfx1250) — matrix operand order repaired and checked by reading, not compiled, not run on hardware
//
#include <hip/hip_runtime.h>

// ---------------------------------------------------------------------------
// CDNA5 (gfx1250) implementation of CrossTransformerBlock3D.
// wave32, WMMA f32_16x16x32_f16 for all dense matmuls + attention.
// Shapes fixed by reference: B=1, D=32, H=64, W=64, C=96, C2=192,
// window 4x4x4 -> 2048 windows x 64 tokens, 8 heads x hd=12.
// ---------------------------------------------------------------------------

typedef __attribute__((ext_vector_type(16))) _Float16 v16h;
typedef __attribute__((ext_vector_type(8)))  _Float16 h8;
typedef __attribute__((ext_vector_type(8)))  float    v8f;

#define NVOX   131072       // 32*64*64
#define DD     32
#define HH     64
#define WW     64
#define C1     96
#define C2     192
#define NWIN   2048
#define NTOK   64
#define NHEAD  8
#define HD     12

// ---------------- fragment loaders (contiguous k order, same for A and B) --------

// A (16x32, row-major source, lda in halves). lane: row = m0+(lane&15),
// element e -> K = kb + hi*16 + e
__device__ __forceinline__ v16h load_a_frag(const _Float16* A, int lda,
                                            int row, int kb, int hi) {
  const _Float16* p = A + (size_t)row * lda + kb + hi * 16;
  h8 lo = *(const h8*)p;
  h8 hv = *(const h8*)(p + 8);
  v16h a;
#pragma unroll
  for (int i = 0; i < 8; ++i) { a[i] = lo[i]; a[i + 8] = hv[i]; }
  return a;
}

// B (32x16). Source is N x K row-major (== weight (out,in) row-major, computes
// X @ W^T).  lane: col = n0+(lane&15), element e -> K = kb + hi*16 + e
__device__ __forceinline__ v16h load_b_frag(const _Float16* B, int ldb,
                                            int col, int kb, int hi) {
  const _Float16* p = B + (size_t)col * ldb + kb + hi * 16;
  h8 lo = *(const h8*)p;
  h8 hv = *(const h8*)(p + 8);
  v16h b;
#pragma unroll
  for (int i = 0; i < 8; ++i) { b[i] = lo[i]; b[i + 8] = hv[i]; }
  return b;
}

// ---------------- small utility kernels -------------------------------------

__global__ void k_f32_to_f16(const float* __restrict__ s,
                             _Float16* __restrict__ d, int n) {
  int i = blockIdx.x * 256 + threadIdx.x;
  if (i < n) d[i] = (_Float16)s[i];
}

// ---------------- K1: LayerNorm(x) + build fp16 [xn | xa] concat ------------
// one wave per voxel; lanes cover channels (96 = 3*32)
__global__ void k_ln_cat(const float* __restrict__ x,
                         const float* __restrict__ xa,
                         const float* __restrict__ w,
                         const float* __restrict__ b,
                         _Float16* __restrict__ cat) {
  int wave = threadIdx.x >> 5, lane = threadIdx.x & 31;
  size_t vox = (size_t)blockIdx.x * 8 + wave;
  const float* xr = x + vox * C1;
  float v0 = xr[lane], v1 = xr[lane + 32], v2 = xr[lane + 64];
  float s = v0 + v1 + v2;
  float ss = v0 * v0 + v1 * v1 + v2 * v2;
#pragma unroll
  for (int m = 1; m < 32; m <<= 1) {
    s  += __shfl_xor(s,  m, 32);
    ss += __shfl_xor(ss, m, 32);
  }
  float mu  = s * (1.0f / 96.0f);
  float var = ss * (1.0f / 96.0f) - mu * mu;
  float rstd = rsqrtf(var + 1e-5f);
  _Float16* cr = cat + vox * C2;
  cr[lane]      = (_Float16)((v0 - mu) * rstd * w[lane]      + b[lane]);
  cr[lane + 32] = (_Float16)((v1 - mu) * rstd * w[lane + 32] + b[lane + 32]);
  cr[lane + 64] = (_Float16)((v2 - mu) * rstd * w[lane + 64] + b[lane + 64]);
  const float* ar = xa + vox * C1;
  cr[96 + lane]      = (_Float16)ar[lane];
  cr[96 + lane + 32] = (_Float16)ar[lane + 32];
  cr[96 + lane + 64] = (_Float16)ar[lane + 64];
}

// ---------------- generic WMMA GEMM: D = A @ B^T + bias ---------------------
// One wave owns one m-tile (16 rows) x ALL n-tiles: the A fragment is loaded
// once per k-step and reused across TN WMMAs (A traffic cut by TN vs 1-tile).
// MODE 0: store fp32 row-major (stride N)          (proj_q)
// MODE 1: store fp16 window-permuted (N==96)       (q/k/v)
// MODE 2: store fp32 window-reversed to d_out      (output projection)
template <int MODE, int TN, int KSTEPS>
__global__ __launch_bounds__(256) void k_gemm(
    const _Float16* __restrict__ A, int lda,
    const _Float16* __restrict__ B, int ldb,
    const float* __restrict__ bias,
    float* __restrict__ fOut, _Float16* __restrict__ hOut,
    int M, float scale) {
  const int N = TN * 16;
  int wave = threadIdx.x >> 5, lane = threadIdx.x & 31;
  int hi = lane >> 4, ln = lane & 15;
  int tm = blockIdx.x * 8 + wave;
  if (tm >= (M >> 4)) return;            // per-wave uniform -> EXEC stays full
  int rowA = tm * 16 + ln;

  v8f acc[TN];
#pragma unroll
  for (int t = 0; t < TN; ++t) acc[t] = v8f{};

#pragma unroll
  for (int ks = 0; ks < KSTEPS; ++ks) {
    int kb = ks * 32;
    v16h a = load_a_frag(A, lda, rowA, kb, hi);   // loaded once, reused TN x
#pragma unroll
    for (int t = 0; t < TN; ++t) {
      v16h b = load_b_frag(B, ldb, t * 16 + ln, kb, hi);
      acc[t] = __builtin_amdgcn_wmma_f32_16x16x32_f16(false, a, false, b,
                                                      (short)0, acc[t],
                                                      false, false);
    }
  }

#pragma unroll
  for (int t = 0; t < TN; ++t) {
    int col = t * 16 + ln;
    float bv = bias[col];
#pragma unroll
    for (int r = 0; r < 8; ++r) {
      int m = tm * 16 + r + hi * 8;
      float val = (acc[t][r] + bv) * scale;
      if (MODE == 0) {
        fOut[(size_t)m * N + col] = val;
      } else if (MODE == 1) {
        // raster voxel -> (window, token) permuted fp16 store
        int d = m >> 12, h = (m >> 6) & 63, w = m & 63;
        int widx = (d >> 2) * 256 + (h >> 2) * 16 + (w >> 2);
        int tok  = (d & 3) * 16 + (h & 3) * 4 + (w & 3);
        hOut[((size_t)widx * NTOK + tok) * C1 + col] = (_Float16)val;
      } else {
        // window-ordered row -> raster voxel fp32 store (window_reverse)
        int widx = m >> 6, tok = m & 63;
        int wd = widx >> 8, wh = (widx >> 4) & 15, ww2 = widx & 15;
        int d = wd * 4 + (tok >> 4), h = wh * 4 + ((tok >> 2) & 3),
            w = ww2 * 4 + (tok & 3);
        fOut[(((size_t)d << 12) + (h << 6) + w) * C1 + col] = val;
      }
    }
  }
}

// ---------------- K3: depthwise 3x3x3 SAME conv (192 groups) ----------------
__global__ void k_dwconv(const float* __restrict__ in,
                         const float* __restrict__ wt,
                         const float* __restrict__ bias,
                         float* __restrict__ out) {
  size_t idx = (size_t)blockIdx.x * 256 + threadIdx.x;   // NVOX*C2 threads
  int c = (int)(idx % C2);
  size_t vox = idx / C2;
  int w = (int)(vox & 63), h = (int)((vox >> 6) & 63), d = (int)(vox >> 12);
  const float* wc = wt + c * 27;
  float acc = bias[c];
#pragma unroll
  for (int dz = -1; dz <= 1; ++dz) {
    int dd = d + dz; if ((unsigned)dd >= (unsigned)DD) continue;
#pragma unroll
    for (int dy = -1; dy <= 1; ++dy) {
      int hh = h + dy; if ((unsigned)hh >= (unsigned)HH) continue;
#pragma unroll
      for (int dx = -1; dx <= 1; ++dx) {
        int ww2 = w + dx; if ((unsigned)ww2 >= (unsigned)WW) continue;
        acc += wc[(dz + 1) * 9 + (dy + 1) * 3 + (dx + 1)] *
               in[((((size_t)dd << 12) + (hh << 6) + ww2) * C2) + c];
      }
    }
  }
  out[idx] = acc;
}

// ---------------- K4: LN + exact GELU + 192->3 offsets + pos ----------------
__global__ void k_offsets(const float* __restrict__ v,
                          const float* __restrict__ lnw,
                          const float* __restrict__ lnb,
                          const float* __restrict__ pw,
                          float* __restrict__ pos) {
  int wave = threadIdx.x >> 5, lane = threadIdx.x & 31;
  size_t vox = (size_t)blockIdx.x * 8 + wave;
  const float* vr = v + vox * C2;
  float vals[6];
  float s = 0.f, ss = 0.f;
#pragma unroll
  for (int j = 0; j < 6; ++j) {
    vals[j] = vr[lane + 32 * j];
    s += vals[j]; ss += vals[j] * vals[j];
  }
#pragma unroll
  for (int m = 1; m < 32; m <<= 1) {
    s  += __shfl_xor(s,  m, 32);
    ss += __shfl_xor(ss, m, 32);
  }
  float mu  = s * (1.0f / 192.0f);
  float var = ss * (1.0f / 192.0f) - mu * mu;
  float rstd = rsqrtf(var + 1e-5f);
  float p0 = 0.f, p1 = 0.f, p2 = 0.f;
#pragma unroll
  for (int j = 0; j < 6; ++j) {
    int c = lane + 32 * j;
    float y = (vals[j] - mu) * rstd * lnw[c] + lnb[c];
    float g = 0.5f * y * (1.0f + erff(y * 0.70710678118654752f));  // exact GELU
    p0 += g * pw[c]; p1 += g * pw[C2 + c]; p2 += g * pw[2 * C2 + c];
  }
#pragma unroll
  for (int m = 1; m < 32; m <<= 1) {
    p0 += __shfl_xor(p0, m, 32);
    p1 += __shfl_xor(p1, m, 32);
    p2 += __shfl_xor(p2, m, 32);
  }
  if (lane == 0) {
    int d = (int)(vox >> 12), h = (int)((vox >> 6) & 63), w = (int)(vox & 63);
    // offsets*tanh*rng*2 ; rng = [1/Dk,1/Hk,1/Wk] = [1/32,1/64,1/64]
    float oz = tanhf(p0) * (2.0f / 32.0f);
    float oy = tanhf(p1) * (2.0f / 64.0f);
    float ox = tanhf(p2) * (2.0f / 64.0f);
    // reference-point normalization replicated with the source's axis quirk
    float rz = ((d + 0.5f) / 64.0f) * 2.0f - 1.0f;   // /Hk
    float ry = ((h + 0.5f) / 64.0f) * 2.0f - 1.0f;   // /Wk
    float rx = ((w + 0.5f) / 32.0f) * 2.0f - 1.0f;   // /Dk
    pos[vox * 3 + 0] = oz + rz;
    pos[vox * 3 + 1] = oy + ry;
    pos[vox * 3 + 2] = ox + rx;
  }
}

// ---------------- K5: trilinear grid sample of xa -> fp16 -------------------
__global__ void k_gridsample(const float* __restrict__ xa,
                             const float* __restrict__ pos,
                             _Float16* __restrict__ out) {
  int wave = threadIdx.x >> 5, lane = threadIdx.x & 31;
  size_t vox = (size_t)blockIdx.x * 8 + wave;
  float pz = pos[vox * 3 + 0], py = pos[vox * 3 + 1], px = pos[vox * 3 + 2];
  float z = (pz + 1.0f) * 0.5f * DD - 0.5f;
  float y = (py + 1.0f) * 0.5f * HH - 0.5f;
  float x = (px + 1.0f) * 0.5f * WW - 0.5f;
  float zf = floorf(z), yf = floorf(y), xf = floorf(x);
  float wz = z - zf, wy = y - yf, wx = x - xf;
  int z0 = min(max((int)zf, 0), DD - 1), z1 = min(max((int)zf + 1, 0), DD - 1);
  int y0 = min(max((int)yf, 0), HH - 1), y1 = min(max((int)yf + 1, 0), HH - 1);
  int x0 = min(max((int)xf, 0), WW - 1), x1 = min(max((int)xf + 1, 0), WW - 1);
  size_t b000 = (((size_t)z0 << 12) + (y0 << 6) + x0) * C1;
  size_t b001 = (((size_t)z0 << 12) + (y0 << 6) + x1) * C1;
  size_t b010 = (((size_t)z0 << 12) + (y1 << 6) + x0) * C1;
  size_t b011 = (((size_t)z0 << 12) + (y1 << 6) + x1) * C1;
  size_t b100 = (((size_t)z1 << 12) + (y0 << 6) + x0) * C1;
  size_t b101 = (((size_t)z1 << 12) + (y0 << 6) + x1) * C1;
  size_t b110 = (((size_t)z1 << 12) + (y1 << 6) + x0) * C1;
  size_t b111 = (((size_t)z1 << 12) + (y1 << 6) + x1) * C1;
#pragma unroll
  for (int j = 0; j < 3; ++j) {
    int c = lane + 32 * j;
    float c00 = xa[b000 + c] * (1 - wx) + xa[b001 + c] * wx;
    float c01 = xa[b010 + c] * (1 - wx) + xa[b011 + c] * wx;
    float c10 = xa[b100 + c] * (1 - wx) + xa[b101 + c] * wx;
    float c11 = xa[b110 + c] * (1 - wx) + xa[b111 + c] * wx;
    float c0 = c00 * (1 - wy) + c01 * wy;
    float c1 = c10 * (1 - wy) + c11 * wy;
    out[vox * C1 + c] = (_Float16)(c0 * (1 - wz) + c1 * wz);
  }
}

// ---------------- K7: windowed cross attention ------------------------------
// 1 block per window, 8 waves = 8 heads. hd=12 zero-padded to WMMA K.
__global__ __launch_bounds__(256) void k_attn(const _Float16* __restrict__ qh,
                                              const _Float16* __restrict__ kh,
                                              const _Float16* __restrict__ vh,
                                              _Float16* __restrict__ oh) {
  __shared__ _Float16 P[NHEAD][NTOK * NTOK];   // 64 KB (per-head private)
  int head = threadIdx.x >> 5, lane = threadIdx.x & 31;
  int hi = lane >> 4, ln = lane & 15;
  size_t wbase = (size_t)blockIdx.x * NTOK * C1;
  const _Float16* Q = qh + wbase;
  const _Float16* K = kh + wbase;
  const _Float16* V = vh + wbase;
  int co = head * HD;

  // --- A fragments of Q (4 tiles of 16 rows, K = 0..31 with 12 real) ---
  v16h af[4];
#pragma unroll
  for (int tm = 0; tm < 4; ++tm) {
    int t = tm * 16 + ln;
    v16h a;
#pragma unroll
    for (int e = 0; e < 16; ++e) a[e] = (_Float16)0.f;
#pragma unroll
    for (int e = 0; e < 16; ++e) {
      int kk = hi * 16 + e;
      if (kk < HD) a[e] = Q[(size_t)t * C1 + co + kk];
    }
    af[tm] = a;
  }
  // --- B fragments of K (cols = key tokens) ---
  v16h bf[4];
#pragma unroll
  for (int tn = 0; tn < 4; ++tn) {
    int n = tn * 16 + ln;
    v16h b;
#pragma unroll
    for (int e = 0; e < 16; ++e) b[e] = (_Float16)0.f;
    if (hi == 0) {
#pragma unroll
      for (int e = 0; e < HD; ++e) b[e] = K[(size_t)n * C1 + co + e];
    }
    bf[tn] = b;
  }

  // --- S = Q K^T (16 WMMAs), register softmax, P -> LDS fp16 ---
#pragma unroll
  for (int tm = 0; tm < 4; ++tm) {
    v8f s0 = {}, s1 = {}, s2 = {}, s3 = {};
    s0 = __builtin_amdgcn_wmma_f32_16x16x32_f16(false, af[tm], false, bf[0], (short)0, s0, false, false);
    s1 = __builtin_amdgcn_wmma_f32_16x16x32_f16(false, af[tm], false, bf[1], (short)0, s1, false, false);
    s2 = __builtin_amdgcn_wmma_f32_16x16x32_f16(false, af[tm], false, bf[2], (short)0, s2, false, false);
    s3 = __builtin_amdgcn_wmma_f32_16x16x32_f16(false, af[tm], false, bf[3], (short)0, s3, false, false);
#pragma unroll
    for (int r = 0; r < 8; ++r) {
      float x0 = s0[r], x1 = s1[r], x2 = s2[r], x3 = s3[r];
      float mx = fmaxf(fmaxf(x0, x1), fmaxf(x2, x3));
#pragma unroll
      for (int m = 1; m < 16; m <<= 1) mx = fmaxf(mx, __shfl_xor(mx, m, 32));
      float e0 = __expf(x0 - mx), e1 = __expf(x1 - mx);
      float e2 = __expf(x2 - mx), e3 = __expf(x3 - mx);
      float sm = e0 + e1 + e2 + e3;
#pragma unroll
      for (int m = 1; m < 16; m <<= 1) sm += __shfl_xor(sm, m, 32);
      float inv = 1.0f / sm;
      int rr = tm * 16 + r + hi * 8;
      _Float16* Pr = &P[head][rr * NTOK];
      Pr[ln]      = (_Float16)(e0 * inv);
      Pr[16 + ln] = (_Float16)(e1 * inv);
      Pr[32 + ln] = (_Float16)(e2 * inv);
      Pr[48 + ln] = (_Float16)(e3 * inv);
    }
  }

  // --- O = P @ V (8 WMMAs: 4 m-tiles x 2 k-steps), N=12 of 16 cols real ---
#pragma unroll
  for (int tm = 0; tm < 4; ++tm) {
    v8f o = {};
#pragma unroll
    for (int ks = 0; ks < 2; ++ks) {
      v16h ap = load_a_frag(&P[head][0], NTOK, tm * 16 + ln, ks * 32, hi);
      v16h bv;
#pragma unroll
      for (int e = 0; e < 16; ++e) bv[e] = (_Float16)0.f;
      if (ln < HD) {
#pragma unroll
        for (int e = 0; e < 16; ++e) {
          int kk = ks * 32 + hi * 16 + e;     // key-token index (all valid)
          bv[e] = V[(size_t)kk * C1 + co + ln];
        }
      }
      o = __builtin_amdgcn_wmma_f32_16x16x32_f16(false, ap, false, bv,
                                                 (short)0, o, false, false);
    }
#pragma unroll
    for (int r = 0; r < 8; ++r) {
      int t = tm * 16 + r + hi * 8;
      if (ln < HD) oh[wbase + (size_t)t * C1 + co + ln] = (_Float16)o[r];
    }
  }
}

// ---------------------------------------------------------------------------
extern "C" void kernel_launch(void* const* d_in, const int* in_sizes, int n_in,
                              void* d_out, int out_size, void* d_ws,
                              size_t ws_size, hipStream_t stream) {
  (void)in_sizes; (void)n_in; (void)out_size; (void)ws_size;
  const float* x        = (const float*)d_in[0];
  const float* xa       = (const float*)d_in[1];
  const float* ln1_w    = (const float*)d_in[2];
  const float* ln1_b    = (const float*)d_in[3];
  const float* projq_w  = (const float*)d_in[4];
  const float* projq_b  = (const float*)d_in[5];
  const float* off_dw_w = (const float*)d_in[6];
  const float* off_dw_b = (const float*)d_in[7];
  const float* off_ln_w = (const float*)d_in[8];
  const float* off_ln_b = (const float*)d_in[9];
  const float* off_pw_w = (const float*)d_in[10];
  const float* wq = (const float*)d_in[11];
  const float* bq = (const float*)d_in[12];
  const float* wk = (const float*)d_in[13];
  const float* bk = (const float*)d_in[14];
  const float* wv = (const float*)d_in[15];
  const float* bv = (const float*)d_in[16];
  const float* wp = (const float*)d_in[17];
  const float* bp = (const float*)d_in[18];
  float* out = (float*)d_out;

  // ---- workspace layout (256B aligned), with reuse ----
  char* ws = (char*)d_ws;
  size_t off = 0;
  auto carve = [&](size_t bytes) -> char* {
    char* p = ws + off;
    off += (bytes + 255) & ~(size_t)255;
    return p;
  };
  _Float16* cat_h = (_Float16*)carve((size_t)NVOX * C2 * 2);  // [xn|xa] fp16
  float*    v1    = (float*)   carve((size_t)NVOX * C2 * 4);  // proj_q out
  float*    v1c   = (float*)   carve((size_t)NVOX * C2 * 4);  // dwconv out
  float*    pos   = (float*)   carve((size_t)NVOX * 3 * 4);
  _Float16* xash  = (_Float16*)carve((size_t)NVOX * C1 * 2);  // grid-sampled xa
  _Float16* whpq  = (_Float16*)carve((size_t)C2 * C2 * 2);
  _Float16* whq   = (_Float16*)carve((size_t)C1 * C1 * 2);
  _Float16* whk   = (_Float16*)carve((size_t)C1 * C1 * 2);
  _Float16* whv   = (_Float16*)carve((size_t)C1 * C1 * 2);
  _Float16* whp   = (_Float16*)carve((size_t)C1 * C1 * 2);
  // reuse: v1 dead after dwconv, v1c dead after offsets
  _Float16* qh = (_Float16*)v1;                               // window order
  _Float16* kh = qh + (size_t)NVOX * C1;
  _Float16* vh = kh + (size_t)NVOX * C1;
  _Float16* oh = (_Float16*)v1c;                              // attn out

  // ---- 0: weight fp32 -> fp16 ----
  k_f32_to_f16<<<(C2 * C2 + 255) / 256, 256, 0, stream>>>(projq_w, whpq, C2 * C2);
  k_f32_to_f16<<<(C1 * C1 + 255) / 256, 256, 0, stream>>>(wq, whq, C1 * C1);
  k_f32_to_f16<<<(C1 * C1 + 255) / 256, 256, 0, stream>>>(wk, whk, C1 * C1);
  k_f32_to_f16<<<(C1 * C1 + 255) / 256, 256, 0, stream>>>(wv, whv, C1 * C1);
  k_f32_to_f16<<<(C1 * C1 + 255) / 256, 256, 0, stream>>>(wp, whp, C1 * C1);

  // ---- 1: LN + concat -> cat_h ----
  k_ln_cat<<<NVOX / 8, 256, 0, stream>>>(x, xa, ln1_w, ln1_b, cat_h);

  // ---- 2: proj_q GEMM (M=131072, N=K=192) -> v1 fp32 ----
  const int mBlocks = (NVOX / 16) / 8;   // 1024: one m-tile per wave
  k_gemm<0, 12, 6><<<mBlocks, 256, 0, stream>>>(
      cat_h, C2, whpq, C2, projq_b, v1, nullptr, NVOX, 1.0f);

  // ---- 3: depthwise 3x3x3 conv ----
  k_dwconv<<<(size_t)NVOX * C2 / 256, 256, 0, stream>>>(v1, off_dw_w, off_dw_b, v1c);

  // ---- 4: LN + GELU + offsets + reference points -> pos ----
  k_offsets<<<NVOX / 8, 256, 0, stream>>>(v1c, off_ln_w, off_ln_b, off_pw_w, pos);

  // ---- 5: grid sample xa -> xash fp16 ----
  k_gridsample<<<NVOX / 8, 256, 0, stream>>>(xa, pos, xash);

  // ---- 6: q/k/v GEMMs (96x96) with window-permuted fp16 stores ----
  k_gemm<1, 6, 3><<<mBlocks, 256, 0, stream>>>(cat_h, C2, whq, C1, bq, nullptr,
                                               qh, NVOX, 0.28867513459481287f);
  k_gemm<1, 6, 3><<<mBlocks, 256, 0, stream>>>(xash, C1, whk, C1, bk, nullptr,
                                               kh, NVOX, 1.0f);
  k_gemm<1, 6, 3><<<mBlocks, 256, 0, stream>>>(cat_h, C2, whv, C1, bv, nullptr,
                                               vh, NVOX, 1.0f);

  // ---- 7: windowed attention ----
  k_attn<<<NWIN, 256, 0, stream>>>(qh, kh, vh, oh);

  // ---- 8: output projection with fused window_reverse ----
  k_gemm<2, 6, 3><<<mBlocks, 256, 0, stream>>>(oh, C1, whp, C1, bp, out,
                                               nullptr, NVOX, 1.0f);
}
